// MultiHeadAttention_37349035606588
// MI455X (gfx1250) — matrix operand order repaired and checked by reading, not compiled, not run on hardware
//
#include <hip/hip_runtime.h>

typedef __bf16 bf16_t;
typedef __attribute__((ext_vector_type(16))) __bf16 v16bf;
typedef __attribute__((ext_vector_type(8)))  __bf16 v8bf;
typedef __attribute__((ext_vector_type(8)))  float  v8f;

#define B_   8
#define S_   512
#define E_   768
#define H_   12
#define HE_  (H_ * E_)   // 9216
#define BH_  (B_ * H_)   // 96
#define SMROWB 2048      // bytes per score-row slot (512 f32, reused as 512 bf16)
#define SMLD   1024      // bf16-element stride of a slot row

// ---- WMMA fragment loaders (all 16B-aligned contiguous loads) ----------
// A 16x32 bf16: lane half h -> K = {k+16h .. k+16h+15} (contiguous)
static __device__ __forceinline__ v16bf load_a(const bf16_t* __restrict__ base,
                                               int ld, int row, int k, int half) {
  const bf16_t* p = base + (size_t)row * ld + k + (half << 4);
  union { v16bf f; v8bf h[2]; } u;
  u.h[0] = *(const v8bf*)p;
  u.h[1] = *(const v8bf*)(p + 8);
  return u.f;
}
// B 32x16 bf16: lane half h -> K = {k+16h .. k+16h+15} (contiguous)
static __device__ __forceinline__ v16bf load_b(const bf16_t* __restrict__ base,
                                               int ld, int row, int k, int half) {
  const bf16_t* p = base + (size_t)row * ld + k + (half << 4);
  union { v16bf f; v8bf h[2]; } u;
  u.h[0] = *(const v8bf*)p;
  u.h[1] = *(const v8bf*)(p + 8);
  return u.f;
}
static __device__ __forceinline__ v8f wmma_bf16(v16bf a, v16bf b, v8f c) {
  return __builtin_amdgcn_wmma_f32_16x16x32_bf16(false, a, false, b, (short)0, c,
                                                 false, false);
}

// 32x64 tile per wave, fully unrolled over K chunks: B fragments are reused
// for both M-halves (12 b128 loads per 8 WMMAs). Full unroll lets loads be
// register-allocated directly into WMMA operands (no ping-pong copies) and
// hoisted across the previous chunk's WMMAs by the scheduler.
template <int NCH>
static __device__ __forceinline__ void gemm_tile32x64(
    const bf16_t* __restrict__ A, int lda, int arow,
    const bf16_t* __restrict__ Bm, int ldb, int brow,
    int half, v8f acc0[4], v8f acc1[4]) {
#pragma unroll
  for (int kc = 0; kc < NCH; ++kc) {
    const int k = kc * 32;
    v16bf a0 = load_a(A, lda, arow, k, half);
    v16bf a1 = load_a(A, lda, arow + 16, k, half);
#pragma unroll
    for (int j = 0; j < 4; ++j) {
      v16bf bf = load_b(Bm, ldb, brow + j * 16, k, half);
      acc0[j] = wmma_bf16(a0, bf, acc0[j]);
      acc1[j] = wmma_bf16(a1, bf, acc1[j]);
    }
  }
}

// ---- helpers ------------------------------------------------------------
__global__ void k_cvt(const float* __restrict__ x, bf16_t* __restrict__ o, int n) {
  int i = blockIdx.x * 256 + threadIdx.x;
  if (i < n) o[i] = (bf16_t)x[i];
}

// in: [batch, R, C] f32 -> out: [batch, C, R] bf16   (R, C multiples of 32)
__global__ void k_tr(const float* __restrict__ in, bf16_t* __restrict__ out,
                     int R, int C) {
  __shared__ float tile[32][33];
  const size_t bofs = (size_t)blockIdx.z * R * C;
  in += bofs; out += bofs;
  const int c0 = blockIdx.x * 32, r0 = blockIdx.y * 32;
  for (int i = threadIdx.y; i < 32; i += 8)
    tile[i][threadIdx.x] = in[(size_t)(r0 + i) * C + c0 + threadIdx.x];
  __syncthreads();
  for (int i = threadIdx.y; i < 32; i += 8)
    out[(size_t)(c0 + i) * R + r0 + threadIdx.x] = (bf16_t)tile[threadIdx.x][i];
}

// ---- GEMM 1: P = X*Wh + bh, dual-store P [s,e] and Pt [e,s] -------------
__global__ __launch_bounds__(256) void k_proj(const bf16_t* __restrict__ xbf,
    const bf16_t* __restrict__ WhT, const float* __restrict__ biasH,
    bf16_t* __restrict__ P, bf16_t* __restrict__ Pt) {
  const int lane = threadIdx.x & 31, half = lane >> 4, nrow = lane & 15;
  const int wave = blockIdx.x * 8 + (threadIdx.x >> 5);
  const int bhid = wave / 192;                 // (512/32)*(768/64)=192 tiles per (b,h)
  const int t = wave % 192, tm = t / 12, tn = t % 12;
  const int bb = bhid / H_, hh = bhid % H_;
  const int sBase = tm * 32, cBase = tn * 64;

  const bf16_t* A  = xbf + (size_t)bb * S_ * E_;
  const bf16_t* Bm = WhT + (size_t)hh * E_ * E_;
  v8f acc0[4] = {}, acc1[4] = {};
  gemm_tile32x64<E_ / 32>(A, E_, sBase + nrow, Bm, E_, cBase + nrow, half,
                          acc0, acc1);

  bf16_t* Pbh  = P  + (size_t)bhid * S_ * E_;
  bf16_t* Ptbh = Pt + (size_t)bhid * E_ * S_;
#pragma unroll
  for (int j = 0; j < 4; ++j) {
    const int col = cBase + j * 16 + nrow;
    const float bias = biasH[hh * E_ + col];
    v8bf o0, o1;
#pragma unroll
    for (int r = 0; r < 8; ++r) {
      float v0 = acc0[j][r] + bias;
      float v1 = acc1[j][r] + bias;
      Pbh[(size_t)(sBase + r + 8 * half) * E_ + col]      = (bf16_t)v0;
      Pbh[(size_t)(sBase + 16 + r + 8 * half) * E_ + col] = (bf16_t)v1;
      o0[r] = (bf16_t)v0;
      o1[r] = (bf16_t)v1;
    }
    *(v8bf*)(Ptbh + (size_t)col * S_ + sBase + 8 * half)      = o0;  // b128
    *(v8bf*)(Ptbh + (size_t)col * S_ + sBase + 16 + 8 * half) = o1;  // b128
  }
}

// ---- GEMM 2: scores = P * P^T * (1/sqrt(E)), fp32 into slot buffer ------
__global__ __launch_bounds__(256) void k_scores(const bf16_t* __restrict__ P,
                                                char* __restrict__ SM) {
  const int lane = threadIdx.x & 31, half = lane >> 4, nrow = lane & 15;
  const int wave = blockIdx.x * 8 + (threadIdx.x >> 5);
  const int bhid = wave / 128;                 // (512/32)*(512/64)=128
  const int t = wave % 128, tm = t / 8, tn = t % 8;
  const int qBase = tm * 32, cBase = tn * 64;
  const bf16_t* Pbh = P + (size_t)bhid * S_ * E_;
  v8f acc0[4] = {}, acc1[4] = {};
  gemm_tile32x64<E_ / 32>(Pbh, E_, qBase + nrow, Pbh, E_, cBase + nrow, half,
                          acc0, acc1);

  const float scale = 0.03608439182435161f;    // 1/sqrt(768)
  char* base = SM + (size_t)bhid * S_ * SMROWB;
#pragma unroll
  for (int j = 0; j < 4; ++j) {
    const int col = cBase + j * 16 + nrow;
#pragma unroll
    for (int r = 0; r < 8; ++r) {
      const int q0 = qBase + r + 8 * half;
      *(float*)(base + (size_t)q0 * SMROWB + (size_t)col * 4) = acc0[j][r] * scale;
      *(float*)(base + (size_t)(q0 + 16) * SMROWB + (size_t)col * 4) =
          acc1[j][r] * scale;
    }
  }
}

// ---- masked softmax: fp32 row -> bf16 attn, in place in the slot --------
__global__ __launch_bounds__(256) void k_softmax(char* __restrict__ SM,
                                                 const int* __restrict__ mask) {
  const int lane = threadIdx.x & 31;
  const int row = blockIdx.x * 8 + (threadIdx.x >> 5);  // BH_*S_ rows
  const int bb = row / (H_ * S_);
  const float* rf = (const float*)(SM + (size_t)row * SMROWB);
  const int* mr = mask + bb * S_;
  const float NEG_INF = -__builtin_inff();
  float v[16];
  float mx = NEG_INF;
#pragma unroll
  for (int i = 0; i < 16; ++i) {
    int idx = i * 32 + lane;
    float x = rf[idx];
    if (mr[idx] != 0) x = NEG_INF;             // mask keys where pad==1
    v[i] = x;
    mx = fmaxf(mx, x);
  }
  for (int off = 16; off > 0; off >>= 1) mx = fmaxf(mx, __shfl_xor(mx, off, 32));
  float s = 0.f;
#pragma unroll
  for (int i = 0; i < 16; ++i) { float e = __expf(v[i] - mx); v[i] = e; s += e; }
  for (int off = 16; off > 0; off >>= 1) s += __shfl_xor(s, off, 32);
  const float inv = 1.0f / s;
  bf16_t* rb = (bf16_t*)(SM + (size_t)row * SMROWB);   // safe: loads consumed above
#pragma unroll
  for (int i = 0; i < 16; ++i) rb[i * 32 + lane] = (bf16_t)(v[i] * inv);
}

// ---- GEMM 3: O = attn * P, written directly into concat layout ----------
__global__ __launch_bounds__(256) void k_av(const char* __restrict__ SM,
    const bf16_t* __restrict__ Pt, bf16_t* __restrict__ cat) {
  const int lane = threadIdx.x & 31, half = lane >> 4, nrow = lane & 15;
  const int wave = blockIdx.x * 8 + (threadIdx.x >> 5);
  const int bhid = wave / 192;                 // (512/32)*(768/64)=192
  const int t = wave % 192, tm = t / 12, tn = t % 12;
  const int bb = bhid / H_, hh = bhid % H_;
  const int qBase = tm * 32, cBase = tn * 64;
  const bf16_t* A  = (const bf16_t*)(SM + (size_t)bhid * S_ * SMROWB);  // ld SMLD
  const bf16_t* Bm = Pt + (size_t)bhid * E_ * S_;                       // rows = e
  v8f acc0[4] = {}, acc1[4] = {};
  gemm_tile32x64<S_ / 32>(A, SMLD, qBase + nrow, Bm, S_, cBase + nrow, half,
                          acc0, acc1);

#pragma unroll
  for (int j = 0; j < 4; ++j) {
    const int col = cBase + j * 16 + nrow;
#pragma unroll
    for (int r = 0; r < 8; ++r) {
      const int q0 = qBase + r + 8 * half;
      cat[(size_t)(bb * S_ + q0) * HE_ + hh * E_ + col]      = (bf16_t)acc0[j][r];
      cat[(size_t)(bb * S_ + q0 + 16) * HE_ + hh * E_ + col] = (bf16_t)acc1[j][r];
    }
  }
}

// ---- GEMM 4: Y = cat * Wo + bo (fp32 out) -------------------------------
// Block = 8 waves x 32 rows = 256x64 tile. The shared 64x32 WoT chunk is
// staged into LDS with async global->LDS copies (ASYNCcnt), double-buffered,
// consumed via ds_load_b128 and reused by 16 M-subtiles. Async loads complete
// in order, so waiting asynccnt<=1 after issuing chunk i+1 guarantees chunk i
// has landed.
#define KOUT_CH (HE_ / 32)   // 288 K-chunks (even)
__global__ __launch_bounds__(256) void k_out(const bf16_t* __restrict__ cat,
    const bf16_t* __restrict__ WoT, const float* __restrict__ bo,
    float* __restrict__ y) {
  __shared__ __attribute__((aligned(16))) bf16_t sB[2][64 * 32];  // 2 x 4KB
  const int lane = threadIdx.x & 31, half = lane >> 4, nrow = lane & 15;
  const int w = threadIdx.x >> 5;
  const int mBlk = blockIdx.x / 12, nBlk = blockIdx.x % 12;
  const int rBase = mBlk * 256 + w * 32, cBase = nBlk * 64;

  // per-thread async copy slot: 16 bytes of one WoT row per chunk
  const int trow = threadIdx.x >> 2, tpart = threadIdx.x & 3;      // 64 rows x 4
  const bf16_t* gsrc = WoT + (size_t)(cBase + trow) * HE_ + tpart * 8;
  const unsigned ldst0 = (unsigned)(uintptr_t)&sB[0][0]
                       + (unsigned)(trow * 64 + tpart * 16);
  const unsigned ldst1 = ldst0 + 64 * 32 * 2;                      // +4096 B

  asm volatile("global_load_async_to_lds_b128 %0, %1, off"
               :: "v"(ldst0), "v"(gsrc) : "memory");               // chunk 0

  const int arow = rBase + nrow;
  v8f acc0[4] = {}, acc1[4] = {};
#pragma unroll 1
  for (int kc = 0; kc < KOUT_CH; kc += 2) {
    // ---- even chunk kc from LDS buf0 ----
    asm volatile("global_load_async_to_lds_b128 %0, %1, off"
                 :: "v"(ldst1), "v"(gsrc + (size_t)(kc + 1) * 32) : "memory");
    v16bf a0 = load_a(cat, HE_, arow, kc * 32, half);
    v16bf a1 = load_a(cat, HE_, arow + 16, kc * 32, half);
    asm volatile("s_wait_asynccnt 0x1" ::: "memory");  // chunk kc landed
    __syncthreads();
#pragma unroll
    for (int j = 0; j < 4; ++j) {
      const bf16_t* bp = &sB[0][0] + (j * 16 + nrow) * 32 + half * 16;
      union { v16bf f; v8bf h[2]; } u;
      u.h[0] = *(const v8bf*)bp;                       // ds_load_b128
      u.h[1] = *(const v8bf*)(bp + 8);
      acc0[j] = wmma_bf16(a0, u.f, acc0[j]);
      acc1[j] = wmma_bf16(a1, u.f, acc1[j]);
    }
    __syncthreads();                                   // buf0 free for re-fill
    // ---- odd chunk kc+1 from LDS buf1 ----
    if (kc + 2 < KOUT_CH) {
      asm volatile("global_load_async_to_lds_b128 %0, %1, off"
                   :: "v"(ldst0), "v"(gsrc + (size_t)(kc + 2) * 32) : "memory");
      asm volatile("s_wait_asynccnt 0x1" ::: "memory");  // chunk kc+1 landed
    } else {
      asm volatile("s_wait_asynccnt 0x0" ::: "memory");
    }
    a0 = load_a(cat, HE_, arow, (kc + 1) * 32, half);
    a1 = load_a(cat, HE_, arow + 16, (kc + 1) * 32, half);
    __syncthreads();
#pragma unroll
    for (int j = 0; j < 4; ++j) {
      const bf16_t* bp = &sB[1][0] + (j * 16 + nrow) * 32 + half * 16;
      union { v16bf f; v8bf h[2]; } u;
      u.h[0] = *(const v8bf*)bp;
      u.h[1] = *(const v8bf*)(bp + 8);
      acc0[j] = wmma_bf16(a0, u.f, acc0[j]);
      acc1[j] = wmma_bf16(a1, u.f, acc1[j]);
    }
    __syncthreads();                                   // buf1 free for re-fill
  }

#pragma unroll
  for (int j = 0; j < 4; ++j) {
    const int col = cBase + j * 16 + nrow;
    const float bias = bo[col];
#pragma unroll
    for (int r = 0; r < 8; ++r) {
      const int m0 = rBase + r + 8 * half;
      y[(size_t)m0 * E_ + col]        = acc0[j][r] + bias;
      y[(size_t)(m0 + 16) * E_ + col] = acc1[j][r] + bias;
    }
  }
}

// ---- LayerNorm (population variance, eps=1e-5) --------------------------
__global__ __launch_bounds__(256) void k_ln(const float* __restrict__ y,
    const float* __restrict__ gamma, const float* __restrict__ beta,
    float* __restrict__ out) {
  const int lane = threadIdx.x & 31;
  const int row = blockIdx.x * 8 + (threadIdx.x >> 5);  // B_*S_ rows
  const float* yr = y + (size_t)row * E_;
  float v[24];
  float s = 0.f, sq = 0.f;
#pragma unroll
  for (int i = 0; i < 24; ++i) {
    float x = yr[i * 32 + lane];
    v[i] = x; s += x; sq += x * x;
  }
  for (int off = 16; off > 0; off >>= 1) {
    s  += __shfl_xor(s,  off, 32);
    sq += __shfl_xor(sq, off, 32);
  }
  const float mu   = s * (1.0f / E_);
  const float var  = sq * (1.0f / E_) - mu * mu;
  const float rstd = rsqrtf(var + 1e-5f);
  float* orow = out + (size_t)row * E_;
#pragma unroll
  for (int i = 0; i < 24; ++i) {
    int e = i * 32 + lane;
    orow[e] = (v[i] - mu) * rstd * gamma[e] + beta[e];
  }
}

extern "C" void kernel_launch(void* const* d_in, const int* in_sizes, int n_in,
                              void* d_out, int out_size, void* d_ws, size_t ws_size,
                              hipStream_t stream) {
  (void)in_sizes; (void)n_in; (void)out_size; (void)ws_size;
  const float* x     = (const float*)d_in[0];
  const int*   mask  = (const int*)  d_in[1];
  const float* Wh    = (const float*)d_in[2];
  const float* bh    = (const float*)d_in[3];
  const float* Wo    = (const float*)d_in[4];
  const float* bo    = (const float*)d_in[5];
  const float* gamma = (const float*)d_in[6];
  const float* beta  = (const float*)d_in[7];
  float* out = (float*)d_out;

  char* ws = (char*)d_ws;
  size_t off = 0;
  auto alloc = [&](size_t bytes) {
    void* p = ws + off;
    off += (bytes + 255) & ~(size_t)255;
    return p;
  };
  bf16_t* xbf = (bf16_t*)alloc((size_t)B_ * S_ * E_ * 2);   //  6.3 MB
  bf16_t* WhT = (bf16_t*)alloc((size_t)H_ * E_ * E_ * 2);   // 14.2 MB
  bf16_t* WoT = (bf16_t*)alloc((size_t)HE_ * E_ * 2);       // 14.2 MB
  bf16_t* P   = (bf16_t*)alloc((size_t)BH_ * S_ * E_ * 2);  // 75.5 MB
  bf16_t* Pt  = (bf16_t*)alloc((size_t)BH_ * E_ * S_ * 2);  // 75.5 MB
  char*   SM  = (char*)  alloc((size_t)BH_ * S_ * SMROWB);  // 100.7 MB
  bf16_t* cat = (bf16_t*)alloc((size_t)B_ * S_ * HE_ * 2);  // 75.5 MB
  float*  y   = (float*) alloc((size_t)B_ * S_ * E_ * 4);   // 12.6 MB

  k_cvt<<<(B_ * S_ * E_ + 255) / 256, 256, 0, stream>>>(x, xbf, B_ * S_ * E_);
  k_tr<<<dim3(E_ / 32, E_ / 32, H_), dim3(32, 8), 0, stream>>>(Wh, WhT, E_, E_);
  k_tr<<<dim3(E_ / 32, HE_ / 32, 1), dim3(32, 8), 0, stream>>>(Wo, WoT, HE_, E_);
  k_proj  <<<BH_ * 192 / 8, 256, 0, stream>>>(xbf, WhT, bh, P, Pt);
  k_scores<<<BH_ * 128 / 8, 256, 0, stream>>>(P, SM);
  k_softmax<<<BH_ * S_ / 8, 256, 0, stream>>>(SM, mask);
  k_av    <<<BH_ * 192 / 8, 256, 0, stream>>>(SM, Pt, cat);
  k_out   <<<16 * 12, 256, 0, stream>>>(cat, WoT, bo, y);
  k_ln    <<<B_ * S_ / 8, 256, 0, stream>>>(y, gamma, beta, out);
}
